// FusedDynamicNeRF_79190607004207
// MI455X (gfx1250) — matrix operand order repaired and checked by reading, not compiled, not run on hardware
//
#include <hip/hip_runtime.h>
#include <hip/hip_bf16.h>

typedef __attribute__((ext_vector_type(16))) _Float16 v16h;
typedef __attribute__((ext_vector_type(8)))  float    v8f;

#define PI_F 3.14159265358979323846f

// ---------------------------------------------------------------------------
// LDS ordering fence for same-wave producer->consumer through LDS (wave32,
// no cross-wave sharing, so a dscnt wait + compiler memory barrier suffices).
__device__ __forceinline__ void lds_fence() {
    asm volatile("s_wait_dscnt 0" ::: "memory");
}

// Single-instruction ReLU: v_med3_f32(v, 0, +inf) == clamp to [0, inf).
__device__ __forceinline__ float relu1(float v) {
    return __builtin_amdgcn_fmed3f(v, 0.0f, __builtin_inff());
}

// ---------------------------------------------------------------------------
// Weight pre-swizzle: f32 row-major W[K][N] -> f16 B-fragment stream.
// Fragment f = nt*KC + kc holds B-chunk (k = kc*32.., n = nt*16..) in the
// v_wmma_f32_16x16x32_f16 B operand, contiguous k order: lane l (n = l&15, kbase = 16*(l>>4)),
// halves j=0..15 -> B[kbase+j][n]. Out-of-range (K/N padding) -> 0.
__global__ void prep_weights_kernel(const float* __restrict__ W,
                                    _Float16* __restrict__ dst,
                                    int K, int Nact, int KC, int NT) {
    int idx = blockIdx.x * blockDim.x + threadIdx.x;
    int total = NT * KC * 512;
    if (idx >= total) return;
    int frag   = idx >> 9;
    int within = idx & 511;
    int lane   = within >> 4;
    int j      = within & 15;
    int nt = frag / KC, kc = frag % KC;
    int n = nt * 16 + (lane & 15);
    int k = kc * 32 + ((lane >> 4) * 16) + j;
    float v = (k < K && n < Nact) ? W[(size_t)k * Nact + n] : 0.0f;
    dst[idx] = (_Float16)v;
}

// ---------------------------------------------------------------------------
// Load an A fragment (16x32 f16) for v_wmma_f32_16x16x32_f16 from row-major
// LDS activations. Lane l: m = mbase + (l&15); halves in contiguous k order (as the B stream):
// g = l>>4 : k = 16g + j   (k in chunk kc).
__device__ __forceinline__ v16h load_a_frag(const _Float16* lds, int stride,
                                            int mbase, int kc, int g, int li) {
    const unsigned int* p =
        (const unsigned int*)(lds + (size_t)(mbase + li) * stride + kc * 32);
    union { v16h h; unsigned int u[8]; } r;
    int b0 = 8 * g;               // uint index of k = 16g
#pragma unroll
    for (int i = 0; i < 4; ++i) r.u[i]     = p[b0 + i];        // k = 16g .. 16g+7
#pragma unroll
    for (int i = 0; i < 4; ++i) r.u[4 + i] = p[b0 + 4 + i];    // k = 16g+8 ..
    return r.h;
}

// ---------------------------------------------------------------------------
// One dense layer for a 32-row wave tile: out[32][N] = act(in[32][Kpad] @ W).
// Weights come pre-swizzled from global (L2-resident). Two M-tiles share each
// B fragment. D layout (ISA): lane l -> n = l&15, rows m = 8*(l>>4)+r.
__device__ __forceinline__ void layer_wmma(const _Float16* __restrict__ wfrag,
                                           int KC,
                                           const _Float16* ldsIn, int inStride,
                                           _Float16* ldsOut, int outStride,
                                           int NT, bool relu, int lane) {
    const int g = lane >> 4, li = lane & 15;
    for (int nt = 0; nt < NT; ++nt) {
        v8f acc0 = {}, acc1 = {};
        for (int kc = 0; kc < KC; ++kc) {
            v16h b = *((const v16h*)wfrag + (size_t)(nt * KC + kc) * 32 + lane);
            v16h a0 = load_a_frag(ldsIn, inStride, 0, kc, g, li);
            v16h a1 = load_a_frag(ldsIn, inStride, 16, kc, g, li);
            acc0 = __builtin_amdgcn_wmma_f32_16x16x32_f16(
                false, a0, false, b, (short)0, acc0, false, false);
            acc1 = __builtin_amdgcn_wmma_f32_16x16x32_f16(
                false, a1, false, b, (short)0, acc1, false, false);
        }
        const int n = nt * 16 + li;
#pragma unroll
        for (int r = 0; r < 8; ++r) {
            float v0 = acc0[r], v1 = acc1[r];
            if (relu) { v0 = relu1(v0); v1 = relu1(v1); }
            ldsOut[(size_t)(8 * g + r) * outStride + n]        = (_Float16)v0;
            ldsOut[(size_t)(16 + 8 * g + r) * outStride + n]   = (_Float16)v1;
        }
    }
}

// ---------------------------------------------------------------------------
// Workspace layout (halves): per-layer fragment streams.
enum : int {
    OFF_W1_0  = 0,        // K 80 ->96  (KC3), N128 (NT8) : 24 frags
    OFF_W1_1  = 12288,    // 128 (KC4), 128 (NT8)         : 32
    OFF_W1_O  = 28672,    // 128, 128                     : 32
    OFF_W2_0  = 45056,    // 208->224 (KC7), 128          : 56
    OFF_W2_1  = 73728,    // 128, 128                     : 32
    OFF_W2_2  = 90112,    // 128, 128                     : 32
    OFF_W2_O  = 106496,   // 128 (KC4), 136->144 (NT9)    : 36
    OFF_WC_0  = 124928,   // 152->160 (KC5), 128          : 40
    OFF_WC_O  = 145408,   // 128 (KC4), 3->16 (NT1)       : 4
    WS_HALVES = 147456
};

#define SA 224  // bufA stride (halves): encP(80)+part1(128) = 208, pad 224
#define SB 144  // bufB stride: hidden 128, pad 144
#define SC 160  // bufC stride: hidden 128 / color input 152, pad 160
#define WAVE_HALVES (32 * (SA + SB + SC))   // 16896 halves = 33 KB per wave
#define LDS_BYTES   (8 * WAVE_HALVES * 2)   // 264 KB (<= 320 KB WGP LDS)

__global__ __launch_bounds__(256, 1)
void FusedDynamicNeRF_kernel(const float* __restrict__ x,
                             const _Float16* __restrict__ wf,
                             float* __restrict__ out) {
    extern __shared__ __align__(16) _Float16 smem[];
    const int lane = threadIdx.x & 31;
    const int wave = threadIdx.x >> 5;
    _Float16* bufA = smem + (size_t)wave * WAVE_HALVES;
    _Float16* bufB = bufA + 32 * SA;
    _Float16* bufC = bufB + 32 * SB;
    const long long rowBase = ((long long)blockIdx.x * 8 + wave) * 32;

    // Zero this wave's LDS (makes all pad columns exact zeros).
    {
        float4* z = (float4*)bufA;
#pragma unroll
        for (int i = 0; i < 66; ++i)                 // 66*32 = 2112 float4 = 33 KB
            z[lane + i * 32] = make_float4(0.f, 0.f, 0.f, 0.f);
    }
    lds_fence();

    // Positional encoding (10 freqs, 4 dims) -> bufA cols [0,80).
    // Hardware transcendentals: outputs are rounded to f16 anyway.
    for (int i = lane; i < 32 * 40; i += 32) {       // uniform 40 iters/lane
        int row = i / 40, t = i % 40, dim = t / 10, fr = t % 10;
        float p   = x[(rowBase + row) * 7 + dim];
        float ang = p * (PI_F * (float)(1 << fr));
        bufA[(size_t)row * SA + dim * 20 + fr]      = (_Float16)__sinf(ang);
        bufA[(size_t)row * SA + dim * 20 + 10 + fr] = (_Float16)__cosf(ang);
    }
    lds_fence();

    layer_wmma(wf + OFF_W1_0, 3, bufA, SA, bufB, SB, 8, true,  lane); // h1
    lds_fence();
    layer_wmma(wf + OFF_W1_1, 4, bufB, SB, bufC, SC, 8, true,  lane); // h2
    lds_fence();
    layer_wmma(wf + OFF_W1_O, 4, bufC, SC, bufA + 80, SA, 8, false, lane); // part1 -> cols 80..207
    lds_fence();
    layer_wmma(wf + OFF_W2_0, 7, bufA, SA, bufB, SB, 8, true,  lane); // [encP|part1] @ w2_0
    lds_fence();
    layer_wmma(wf + OFF_W2_1, 4, bufB, SB, bufC, SC, 8, true,  lane);
    lds_fence();
    layer_wmma(wf + OFF_W2_2, 4, bufC, SC, bufB, SB, 8, true,  lane);
    lds_fence();
    layer_wmma(wf + OFF_W2_O, 4, bufB, SB, bufA, SA, 9, false, lane); // dens_out -> bufA[0..136)
    lds_fence();

    // density / scene_flow / disocc (one row per lane; 32 rows per wave).
    {
        const _Float16* d = bufA + (size_t)lane * SA;
        float* o = out + (rowBase + lane) * 12;
        o[3] = (float)d[8];                                   // density
#pragma unroll
        for (int i = 0; i < 6; ++i) o[4 + i] = tanhf((float)d[i]);   // flow
#pragma unroll
        for (int i = 0; i < 2; ++i) {                         // disocc
            float t = (float)d[6 + i];
            o[10 + i] = 1.0f / (1.0f + expf(-t));
        }
    }

    // View encoding (4 freqs, 3 dims) -> bufC cols [0,24).
    for (int i = lane; i < 32 * 12; i += 32) {
        int row = i / 12, t = i % 12, dim = t / 4, fr = t % 4;
        float p   = x[(rowBase + row) * 7 + 4 + dim];
        float ang = p * (PI_F * (float)(1 << fr));
        bufC[(size_t)row * SC + dim * 8 + fr]     = (_Float16)__sinf(ang);
        bufC[(size_t)row * SC + dim * 8 + 4 + fr] = (_Float16)__cosf(ang);
    }
    // feature = dens_out[:, 8:136] -> bufC cols [24,152).
    for (int i = lane; i < 32 * 128; i += 32) {
        int row = i >> 7, c = i & 127;
        bufC[(size_t)row * SC + 24 + c] = bufA[(size_t)row * SA + 8 + c];
    }
    lds_fence();

    layer_wmma(wf + OFF_WC_0, 5, bufC, SC, bufB, SB, 8, true, lane); // color hidden
    lds_fence();

    // rgb = hc @ wc_out : single N-tile, write D fragment straight to global.
    {
        const int g = lane >> 4, li = lane & 15;
        v8f acc0 = {}, acc1 = {};
        for (int kc = 0; kc < 4; ++kc) {
            v16h b = *((const v16h*)(wf + OFF_WC_O) + kc * 32 + lane);
            v16h a0 = load_a_frag(bufB, SB, 0, kc, g, li);
            v16h a1 = load_a_frag(bufB, SB, 16, kc, g, li);
            acc0 = __builtin_amdgcn_wmma_f32_16x16x32_f16(
                false, a0, false, b, (short)0, acc0, false, false);
            acc1 = __builtin_amdgcn_wmma_f32_16x16x32_f16(
                false, a1, false, b, (short)0, acc1, false, false);
        }
        if (li < 3) {
#pragma unroll
            for (int r = 0; r < 8; ++r) {
                out[(rowBase + 8 * g + r) * 12 + li]      = acc0[r];
                out[(rowBase + 16 + 8 * g + r) * 12 + li] = acc1[r];
            }
        }
    }
}

// ---------------------------------------------------------------------------
extern "C" void kernel_launch(void* const* d_in, const int* in_sizes, int n_in,
                              void* d_out, int out_size, void* d_ws, size_t ws_size,
                              hipStream_t stream) {
    (void)in_sizes; (void)n_in; (void)out_size; (void)ws_size;
    const float* x = (const float*)d_in[0];
    _Float16* wf = (_Float16*)d_ws;   // needs 294,912 bytes of workspace

    struct { int widx, K, N, KC, NT, off; } L[9] = {
        {1,  80, 128, 3, 8, OFF_W1_0},
        {2, 128, 128, 4, 8, OFF_W1_1},
        {3, 128, 128, 4, 8, OFF_W1_O},
        {4, 208, 128, 7, 8, OFF_W2_0},
        {5, 128, 128, 4, 8, OFF_W2_1},
        {6, 128, 128, 4, 8, OFF_W2_2},
        {7, 128, 136, 4, 9, OFF_W2_O},
        {8, 152, 128, 5, 8, OFF_WC_0},
        {9, 128,   3, 4, 1, OFF_WC_O},
    };
    for (int i = 0; i < 9; ++i) {
        int total = L[i].NT * L[i].KC * 512;
        int blocks = (total + 255) / 256;
        prep_weights_kernel<<<blocks, 256, 0, stream>>>(
            (const float*)d_in[L[i].widx], wf + L[i].off,
            L[i].K, L[i].N, L[i].KC, L[i].NT);
    }

    const int ROWS_PER_BLOCK = 256;                  // 8 waves * 32 rows
    const int N = 2097152;
    FusedDynamicNeRF_kernel<<<N / ROWS_PER_BLOCK, 256, LDS_BYTES, stream>>>(
        x, wf, (float*)d_out);
}
